// GATppi_55396488184264
// MI455X (gfx1250) — matrix operand order repaired and checked by reading, not compiled, not run on hardware
//
#include <hip/hip_runtime.h>
#include <hip/hip_bf16.h>

#define NN 50000
#define EE 400000
#define NFEAT 256
#define KDIM 256      // inner dim of every GEMM
#define KSTEPS (KDIM / 32)
#define HH 4          // heads, all layers
#define NHID 64
#define NCLASS 121

typedef __attribute__((ext_vector_type(16))) __bf16 v16bf;
typedef __attribute__((ext_vector_type(8)))  __bf16 v8bf;
typedef __attribute__((ext_vector_type(8)))  float  v8f;

static __device__ __forceinline__ __bf16 f2bf(float f) {
    unsigned u = __builtin_bit_cast(unsigned, f);
    unsigned r = (u + 0x7FFFu + ((u >> 16) & 1u)) >> 16;   // RNE
    unsigned short s = (unsigned short)r;
    return __builtin_bit_cast(__bf16, s);
}

// ---------------- elementwise conversions ----------------

__global__ void cvt_bf16_kernel(const float* __restrict__ in, __bf16* __restrict__ out, int n) {
    int t = blockIdx.x * blockDim.x + threadIdx.x;
    if (t < n) out[t] = f2bf(in[t]);
}

__global__ void elu_bf16_kernel(const float* __restrict__ in, __bf16* __restrict__ out, int n) {
    int t = blockIdx.x * blockDim.x + threadIdx.x;
    if (t < n) {
        float v = in[t];
        float e = v > 0.f ? v : expm1f(v);
        out[t] = f2bf(e);
    }
}

// ---------------- W packing into WMMA B-fragment layout ----------------
// wpk element layout: [colTile][kstep][lane][e], e = 0..15 per lane (32 bytes)
// lane holds column n = colTile*16 + (lane&15); K index = kstep*32 + (lane>>4)*8 + (e&7) + (e>>3)*16
__global__ void pack_w_kernel(const float* __restrict__ W, __bf16* __restrict__ wpk,
                              int ncol, int total) {
    int idx = blockIdx.x * blockDim.x + threadIdx.x;
    if (idx >= total) return;
    int e    = idx & 15;
    int lane = (idx >> 4) & 31;
    int ks   = (idx >> 9) & 7;
    int ct   = idx >> 12;
    int n = ct * 16 + (lane & 15);
    int k = ks * 32 + ((lane >> 4) * 8) + (e & 7) + ((e >> 3) * 16);
    float v = (n < ncol) ? W[(size_t)k * ncol + n] : 0.f;
    wpk[idx] = f2bf(v);
}

// ---------------- GEMM: feat[N, ldf] = xb[N, 256] @ W (via packed wpk) ----------------
// One wave computes one 16x16 output tile; K=256 -> 8 x v_wmma_f32_16x16x32_bf16
__global__ void gemm_bf16_kernel(const __bf16* __restrict__ xb,
                                 const __bf16* __restrict__ wpk,
                                 float* __restrict__ feat,
                                 int ldf, int colTiles) {
    int lane = threadIdx.x & 31;
    int wid  = threadIdx.x >> 5;
    int ct   = blockIdx.y * 4 + wid;          // uniform within wave
    int rt   = blockIdx.x;
    if (ct >= colTiles) return;               // whole-wave uniform exit (EXEC all-1 for WMMA)

    // A fragment source: row m = lane&15, K chunks at (lane>>4)*8 and +16
    const __bf16* arow = xb + (size_t)(rt * 16 + (lane & 15)) * KDIM + ((lane >> 4) * 8);
    const __bf16* bbase = wpk + ((size_t)ct * KSTEPS * 32 + lane) * 16;

    v8f c = {};
#pragma unroll
    for (int ks = 0; ks < KSTEPS; ++ks) {
        union { v16bf v; v8bf h[2]; } a;
        a.h[0] = *(const v8bf*)(arow + ks * 32);
        a.h[1] = *(const v8bf*)(arow + ks * 32 + 16);
        v16bf b = *(const v16bf*)(bbase + (size_t)ks * 32 * 16);
        c = __builtin_amdgcn_wmma_f32_16x16x32_bf16(false, a.v, false, b,
                                                    (short)0, c, false, false);
    }
    int col   = ct * 16 + (lane & 15);
    int rbase = rt * 16 + ((lane >> 4) << 3);
#pragma unroll
    for (int r = 0; r < 8; ++r)
        feat[(size_t)(rbase + r) * ldf + col] = c[r];
}

// ---------------- attention scores: el/er[n,h] = sum_d feat[n,h,d]*al/ar[h,d] ----------------
__global__ void attn_scores_kernel(const float* __restrict__ feat,
                                   const float* __restrict__ al, const float* __restrict__ ar,
                                   float* __restrict__ el, float* __restrict__ er,
                                   int nh, int D, int ldf) {
    int w = (blockIdx.x * blockDim.x + threadIdx.x) >> 5;
    int lane = threadIdx.x & 31;
    if (w >= nh) return;
    int n = w / HH, h = w - n * HH;
    const float* f = feat + (size_t)n * ldf + h * D;
    const float* a = al + h * D;
    const float* b = ar + h * D;
    float s1 = 0.f, s2 = 0.f;
    for (int d = lane; d < D; d += 32) {
        float v = f[d];
        s1 += v * a[d];
        s2 += v * b[d];
    }
#pragma unroll
    for (int off = 16; off > 0; off >>= 1) {
        s1 += __shfl_xor(s1, off, 32);
        s2 += __shfl_xor(s2, off, 32);
    }
    if (lane == 0) { el[w] = s1; er[w] = s2; }
}

// ---------------- edge logits + segment max (monotone uint key) ----------------
static __device__ __forceinline__ unsigned fkey(float f) {
    unsigned u = __builtin_bit_cast(unsigned, f);
    return (u & 0x80000000u) ? ~u : (u | 0x80000000u);
}
static __device__ __forceinline__ float keyf(unsigned k) {
    unsigned u = (k & 0x80000000u) ? (k & 0x7FFFFFFFu) : ~k;
    return __builtin_bit_cast(float, u);
}

__global__ void edge_logits_kernel(const int* __restrict__ src, const int* __restrict__ dst,
                                   const float* __restrict__ el, const float* __restrict__ er,
                                   float* __restrict__ ebuf, unsigned* __restrict__ mkey,
                                   int eh) {
    int t = blockIdx.x * blockDim.x + threadIdx.x;
    if (t >= eh) return;
    int i = t / HH, h = t - i * HH;
    int s = src[i], d = dst[i];
    float v = el[(size_t)s * HH + h] + er[(size_t)d * HH + h];
    v = v > 0.f ? v : 0.2f * v;          // leaky relu
    ebuf[t] = v;
    atomicMax(mkey + (size_t)d * HH + h, fkey(v));
}

__global__ void edge_expsum_kernel(const int* __restrict__ dst,
                                   float* __restrict__ ebuf,
                                   const unsigned* __restrict__ mkey,
                                   float* __restrict__ ssum, int eh) {
    int t = blockIdx.x * blockDim.x + threadIdx.x;
    if (t >= eh) return;
    int i = t / HH, h = t - i * HH;
    int d = dst[i];
    float m = keyf(mkey[(size_t)d * HH + h]);
    float ex = __expf(ebuf[t] - m);
    ebuf[t] = ex;
    atomicAdd(ssum + (size_t)d * HH + h, ex);
}

// ---------------- message aggregation: one wave per (edge, head) ----------------
__global__ void aggregate_kernel(const int* __restrict__ src, const int* __restrict__ dst,
                                 const float* __restrict__ ebuf, const float* __restrict__ ssum,
                                 const float* __restrict__ feat, float* __restrict__ agg,
                                 int eh, int D, int ldf, int lda) {
    int w = (blockIdx.x * blockDim.x + threadIdx.x) >> 5;
    int lane = threadIdx.x & 31;
    if (w >= eh) return;
    int i = w / HH, h = w - i * HH;
    int sn = src[i], dn = dst[i];
    float alpha = ebuf[w] / ssum[(size_t)dn * HH + h];
    const float* f = feat + (size_t)sn * ldf + h * D;
    float* o = agg + (size_t)dn * lda + h * D;
    for (int d = lane; d < D; d += 32)
        atomicAdd(o + d, alpha * f[d]);
}

// ---------------- output: mean over heads ----------------
__global__ void head_mean_kernel(const float* __restrict__ agg, float* __restrict__ out, int nc) {
    int t = blockIdx.x * blockDim.x + threadIdx.x;
    if (t >= nc) return;
    int n = t / NCLASS, c = t - n * NCLASS;
    const float* a = agg + (size_t)n * (HH * NCLASS);
    out[t] = 0.25f * (a[c] + a[NCLASS + c] + a[2 * NCLASS + c] + a[3 * NCLASS + c]);
}

// =====================================================================

extern "C" void kernel_launch(void* const* d_in, const int* in_sizes, int n_in,
                              void* d_out, int out_size, void* d_ws, size_t ws_size,
                              hipStream_t stream) {
    const float* h   = (const float*)d_in[0];
    const int*   src = (const int*)d_in[1];
    const int*   dst = (const int*)d_in[2];
    const float* Ws[3]  = { (const float*)d_in[3], (const float*)d_in[6], (const float*)d_in[9]  };
    const float* als[3] = { (const float*)d_in[4], (const float*)d_in[7], (const float*)d_in[10] };
    const float* ars[3] = { (const float*)d_in[5], (const float*)d_in[8], (const float*)d_in[11] };

    // ---- workspace carve ----
    char* ws = (char*)d_ws;
    size_t off = 0;
    auto carve = [&](size_t bytes) -> void* {
        void* p = ws + off;
        off = (off + bytes + 255) & ~(size_t)255;
        return p;
    };
    __bf16*   xb   = (__bf16*)  carve((size_t)NN * 256 * 2);
    float*    feat = (float*)   carve((size_t)NN * 496 * 4);
    float*    agg  = (float*)   carve((size_t)NN * 496 * 4);
    __bf16*   wpk  = (__bf16*)  carve((size_t)31 * 8 * 32 * 16 * 2);
    float*    el   = (float*)   carve((size_t)NN * HH * 4);
    float*    er   = (float*)   carve((size_t)NN * HH * 4);
    unsigned* mkey = (unsigned*)carve((size_t)NN * HH * 4);
    float*    ssum = (float*)   carve((size_t)NN * HH * 4);
    float*    ebuf = (float*)   carve((size_t)EE * HH * 4);

    const int rowTiles = NN / 16;                 // 3125 exactly
    const int nh = NN * HH;
    const int eh = EE * HH;

    // layer params: Ncol (real), colTiles (padded), D, ldf, lda(agg stride)
    const int L_ncol[3] = { 256, 256, 484 };
    const int L_ct[3]   = { 16, 16, 31 };
    const int L_D[3]    = { NHID, NHID, NCLASS };
    const int L_ldf[3]  = { 256, 256, 496 };
    const int L_lda[3]  = { 256, 256, 484 };

    // initial activation -> bf16
    {
        int n = NN * NFEAT;
        cvt_bf16_kernel<<<(n + 255) / 256, 256, 0, stream>>>(h, xb, n);
    }

    for (int L = 0; L < 3; ++L) {
        const int ncol = L_ncol[L], cts = L_ct[L], D = L_D[L];
        const int ldf = L_ldf[L], lda = L_lda[L];

        // pack W into B-fragment order (zero-padded columns for layer 3)
        int wtotal = cts * 8 * 32 * 16;
        pack_w_kernel<<<(wtotal + 255) / 256, 256, 0, stream>>>(Ws[L], wpk, ncol, wtotal);

        // zero accumulators
        hipMemsetAsync(agg, 0, (size_t)NN * lda * 4, stream);
        hipMemsetAsync(mkey, 0, (size_t)NN * HH * 4, stream);
        hipMemsetAsync(ssum, 0, (size_t)NN * HH * 4, stream);

        // GEMM: feat = xb @ W   (bf16 WMMA, fp32 accum)
        dim3 ggrid(rowTiles, (cts + 3) / 4);
        gemm_bf16_kernel<<<ggrid, 128, 0, stream>>>(xb, wpk, feat, ldf, cts);

        // attention scores (wave per (node,head))
        {
            int waves = nh;
            int blocks = (waves * 32 + 255) / 256;
            attn_scores_kernel<<<blocks, 256, 0, stream>>>(feat, als[L], ars[L], el, er, nh, D, ldf);
        }

        // edge logits + segment max
        edge_logits_kernel<<<(eh + 255) / 256, 256, 0, stream>>>(src, dst, el, er, ebuf, mkey, eh);

        // exp + segment sum
        edge_expsum_kernel<<<(eh + 255) / 256, 256, 0, stream>>>(dst, ebuf, mkey, ssum, eh);

        // weighted scatter-add (wave per (edge,head))
        {
            int blocks = (eh * 32 + 255) / 256;
            aggregate_kernel<<<blocks, 256, 0, stream>>>(src, dst, ebuf, ssum, feat, agg,
                                                         eh, D, ldf, lda);
        }

        if (L < 2) {
            int n = NN * 256;
            elu_bf16_kernel<<<(n + 255) / 256, 256, 0, stream>>>(agg, xb, n);
        } else {
            int n = NN * NCLASS;
            head_mean_kernel<<<(n + 255) / 256, 256, 0, stream>>>(agg, (float*)d_out, n);
        }
    }
}
